// SwiGLUMLP_6640019440343
// MI455X (gfx1250) — matrix operand order repaired and checked by reading, not compiled, not run on hardware
//
#include <hip/hip_runtime.h>
#include <hip/hip_bf16.h>

typedef __attribute__((ext_vector_type(16))) _Float16 v16h;
typedef __attribute__((ext_vector_type(8)))  _Float16 h8;
typedef __attribute__((ext_vector_type(4)))  _Float16 h4;
typedef __attribute__((ext_vector_type(8)))  float    v8f;
typedef __attribute__((ext_vector_type(4)))  float    f4;

// Octonion inverse tables: Big[i*din+r][k*dout+c] = SSGN[i][k]*wq[JINV[i][k]][r][c]
__constant__ int JINV[8][8] = {
  {0,1,2,3,4,5,6,7},
  {1,0,3,2,5,4,7,6},
  {2,3,0,1,6,7,4,5},
  {3,2,1,0,7,6,5,4},
  {4,5,6,7,0,1,2,3},
  {5,4,7,6,1,0,3,2},
  {6,7,4,5,2,3,0,1},
  {7,6,5,4,3,2,1,0}};
__constant__ float SSGN[8][8] = {
  {+1,+1,+1,+1,+1,+1,+1,+1},
  {-1,+1,-1,+1,-1,+1,+1,-1},
  {-1,+1,+1,-1,-1,-1,+1,+1},
  {-1,-1,+1,+1,-1,+1,-1,+1},
  {-1,+1,+1,+1,+1,-1,-1,-1},
  {-1,-1,+1,-1,+1,+1,+1,-1},
  {-1,-1,-1,+1,+1,-1,+1,+1},
  {-1,+1,-1,-1,+1,+1,-1,+1}};

// ---------------- scale = mean(|w|) + 1e-8 (one block per tensor) ----------------
__global__ void oct_absmean(const float* __restrict__ w0, const float* __restrict__ w1,
                            const float* __restrict__ w2, float* __restrict__ scales, int n) {
  const float* w = (blockIdx.x == 0) ? w0 : (blockIdx.x == 1 ? w1 : w2);
  __shared__ float red[256];
  float s = 0.f;
  for (int i = threadIdx.x; i < n; i += 256) s += fabsf(w[i]);
  red[threadIdx.x] = s;
  __syncthreads();
  for (int off = 128; off > 0; off >>= 1) {
    if (threadIdx.x < off) red[threadIdx.x] += red[threadIdx.x + off];
    __syncthreads();
  }
  if (threadIdx.x == 0) scales[blockIdx.x] = red[0] / (float)n + 1e-8f;
}

// ---------------- x (f32) -> f16, 4 elements/thread ----------------
__global__ void oct_cvt_f16(const float* __restrict__ x, _Float16* __restrict__ xh) {
  long i = ((long)blockIdx.x * blockDim.x + threadIdx.x) * 4;
  f4 v = *(const f4*)(x + i);
  h4 o;
  o[0] = (_Float16)v[0]; o[1] = (_Float16)v[1];
  o[2] = (_Float16)v[2]; o[3] = (_Float16)v[3];
  *(h4*)(xh + i) = o;
}

// ---------------- build transposed octonion-expanded ternary matrix ----------------
__global__ void oct_build_bigT(const float* __restrict__ w, const float* __restrict__ scales,
                               int scale_idx, _Float16* __restrict__ bigT, int din8, int dout8) {
  int Ktot = 8 * din8;
  long idx = (long)blockIdx.x * blockDim.x + threadIdx.x;
  int kk = (int)(idx % Ktot);
  int nn = (int)(idx / Ktot);
  int i = kk / din8, r = kk % din8;
  int kb = nn / dout8, c = nn % dout8;
  int j = JINV[i][kb];
  float sg = SSGN[i][kb];
  float inv = 1.0f / scales[scale_idx];
  float wv = w[((long)j * din8 + r) * dout8 + c];
  float q = rintf(wv * inv);
  q = fminf(1.f, fmaxf(-1.f, q));
  bigT[idx] = (_Float16)(sg * q);
}

// ---------------- WMMA fragment helpers ----------------
__device__ __forceinline__ v16h load_afrag(const _Float16* As, int lds, int row, int kh) {
  h8 lo = *(const h8*)&As[row * lds + kh];
  h8 hi = *(const h8*)&As[row * lds + kh + 8];
  return __builtin_shufflevector(lo, hi, 0,1,2,3,4,5,6,7,8,9,10,11,12,13,14,15);
}
__device__ __forceinline__ v16h load_bfrag(const _Float16* Bs, int lds, int row, int kh) {
  h8 lo = *(const h8*)&Bs[row * lds + kh];
  h8 hi = *(const h8*)&Bs[row * lds + kh + 8];
  return __builtin_shufflevector(lo, hi, 0,1,2,3,4,5,6,7,8,9,10,11,12,13,14,15);
}
__device__ __forceinline__ v8f wmma_f16(v16h a, v16h b, v8f c) {
  return __builtin_amdgcn_wmma_f32_16x16x32_f16(false, a, false, b, (short)0, c, false, false);
}

// ---------------- GEMM1 (fused gate + up + SwiGLU) ----------------
// BM=128, BN=128, BK=64 double-buffered; 8 waves as 4(M)x2(N); wave tile 32x64.
// A = xh [M][768] f16; BgT,BuT = [2048][768] f16 (N-major); Hout = [M][2048] f16
__launch_bounds__(256)
__global__ void oct_gemm_gateup(const _Float16* __restrict__ A,
                                const _Float16* __restrict__ BgT,
                                const _Float16* __restrict__ BuT,
                                const float* __restrict__ scales,
                                _Float16* __restrict__ Hout) {
  constexpr int Kd = 768, Nd = 2048, BM = 128, BN = 128, BK = 64, LDSP = BK + 8;
  __shared__ __align__(16) _Float16 As[2][BM * LDSP];
  __shared__ __align__(16) _Float16 Bg[2][BN * LDSP];
  __shared__ __align__(16) _Float16 Bu[2][BN * LDSP];

  const int tid = threadIdx.x, lane = tid & 31, wave = tid >> 5;
  const int waveM = wave & 3, waveN = wave >> 2;
  const long m0 = (long)blockIdx.y * BM;
  const int  n0 = blockIdx.x * BN;
  const int fr  = lane & 15;
  const int hiA = (lane >> 4) * 8;
  const int hiB = (lane >> 4) * 16;

  // global->reg staging: 4 h8 chunks each for A / Bg / Bu (tile rows = 128)
  h8 ra[4], rg[4], ru[4];
  const int gRow = tid >> 3, gCo = (tid & 7) * 8;   // +32 rows per chunk

  auto loadG = [&](int kt) {
    #pragma unroll
    for (int t = 0; t < 4; t++) {
      ra[t] = *(const h8*)&A[(m0 + gRow + t * 32) * Kd + kt + gCo];
      rg[t] = *(const h8*)&BgT[(long)(n0 + gRow + t * 32) * Kd + kt + gCo];
      ru[t] = *(const h8*)&BuT[(long)(n0 + gRow + t * 32) * Kd + kt + gCo];
    }
  };
  auto storeL = [&](int buf) {
    #pragma unroll
    for (int t = 0; t < 4; t++) {
      *(h8*)&As[buf][(gRow + t * 32) * LDSP + gCo] = ra[t];
      *(h8*)&Bg[buf][(gRow + t * 32) * LDSP + gCo] = rg[t];
      *(h8*)&Bu[buf][(gRow + t * 32) * LDSP + gCo] = ru[t];
    }
  };

  v8f accg[2][4] = {}, accu[2][4] = {};

  loadG(0);
  storeL(0);
  __syncthreads();
  int cur = 0;

  for (int kt = 0; kt < Kd; kt += BK) {
    const bool has_next = (kt + BK) < Kd;
    if (has_next) loadG(kt + BK);   // overlap next-tile global loads with compute

    const _Float16* Ac  = As[cur];
    const _Float16* Bgc = Bg[cur];
    const _Float16* Buc = Bu[cur];
    #pragma unroll
    for (int ks = 0; ks < 2; ks++) {
      const int kh = ks * 32;
      v16h af[2], bgf[4], buf_[4];
      #pragma unroll
      for (int mi = 0; mi < 2; mi++)
        af[mi] = load_afrag(Ac, LDSP, waveM * 32 + mi * 16 + fr, kh + hiB);
      #pragma unroll
      for (int ni = 0; ni < 4; ni++) {
        bgf[ni]  = load_bfrag(Bgc, LDSP, waveN * 64 + ni * 16 + fr, kh + hiB);
        buf_[ni] = load_bfrag(Buc, LDSP, waveN * 64 + ni * 16 + fr, kh + hiB);
      }
      #pragma unroll
      for (int mi = 0; mi < 2; mi++)
        #pragma unroll
        for (int ni = 0; ni < 4; ni++) {
          accg[mi][ni] = wmma_f16(af[mi], bgf[ni],  accg[mi][ni]);
          accu[mi][ni] = wmma_f16(af[mi], buf_[ni], accu[mi][ni]);
        }
    }

    if (has_next) {
      storeL(cur ^ 1);
      __syncthreads();
      cur ^= 1;
    }
  }

  const float sg = scales[0], su = scales[1];
  const int nlane = lane & 15;
  const int mlane = (lane >> 4) * 8;
  #pragma unroll
  for (int mi = 0; mi < 2; mi++)
    #pragma unroll
    for (int ni = 0; ni < 4; ni++) {
      long mbase = m0 + waveM * 32 + mi * 16 + mlane;
      int  nn = n0 + waveN * 64 + ni * 16 + nlane;
      #pragma unroll
      for (int e = 0; e < 8; e++) {
        float g = sg * accg[mi][ni][e];
        float u = su * accu[mi][ni][e];
        float sig = __builtin_amdgcn_rcpf(1.0f + __expf(-g));  // fast sigmoid
        Hout[(mbase + e) * (long)Nd + nn] = (_Float16)(g * sig * u);
      }
    }
}

// ---------------- GEMM2 (down projection) ----------------
// BM=128, BN=128, BK=64 double-buffered; wave tile 32x64.
// A = h [M][2048] f16; BdT = [768][2048] f16 (N-major); out fp32 [M][768]
__launch_bounds__(256)
__global__ void oct_gemm_down(const _Float16* __restrict__ A,
                              const _Float16* __restrict__ BdT,
                              const float* __restrict__ scales,
                              float* __restrict__ Out) {
  constexpr int Kd = 2048, Nd = 768, BM = 128, BN = 128, BK = 64, LDSP = BK + 8;
  __shared__ __align__(16) _Float16 As[2][BM * LDSP];
  __shared__ __align__(16) _Float16 Bs[2][BN * LDSP];

  const int tid = threadIdx.x, lane = tid & 31, wave = tid >> 5;
  const int waveM = wave & 3, waveN = wave >> 2;
  const long m0 = (long)blockIdx.y * BM;
  const int  n0 = blockIdx.x * BN;
  const int fr  = lane & 15;
  const int hiA = (lane >> 4) * 8;
  const int hiB = (lane >> 4) * 16;

  h8 ra[4], rb[4];
  const int gRow = tid >> 3, gCo = (tid & 7) * 8;

  auto loadG = [&](int kt) {
    #pragma unroll
    for (int t = 0; t < 4; t++) {
      ra[t] = *(const h8*)&A[(m0 + gRow + t * 32) * Kd + kt + gCo];
      rb[t] = *(const h8*)&BdT[(long)(n0 + gRow + t * 32) * Kd + kt + gCo];
    }
  };
  auto storeL = [&](int buf) {
    #pragma unroll
    for (int t = 0; t < 4; t++) {
      *(h8*)&As[buf][(gRow + t * 32) * LDSP + gCo] = ra[t];
      *(h8*)&Bs[buf][(gRow + t * 32) * LDSP + gCo] = rb[t];
    }
  };

  v8f acc[2][4] = {};

  loadG(0);
  storeL(0);
  __syncthreads();
  int cur = 0;

  for (int kt = 0; kt < Kd; kt += BK) {
    const bool has_next = (kt + BK) < Kd;
    if (has_next) loadG(kt + BK);

    const _Float16* Ac = As[cur];
    const _Float16* Bc = Bs[cur];
    #pragma unroll
    for (int ks = 0; ks < 2; ks++) {
      const int kh = ks * 32;
      v16h af[2], bf[4];
      #pragma unroll
      for (int mi = 0; mi < 2; mi++)
        af[mi] = load_afrag(Ac, LDSP, waveM * 32 + mi * 16 + fr, kh + hiB);
      #pragma unroll
      for (int ni = 0; ni < 4; ni++)
        bf[ni] = load_bfrag(Bc, LDSP, waveN * 64 + ni * 16 + fr, kh + hiB);
      #pragma unroll
      for (int mi = 0; mi < 2; mi++)
        #pragma unroll
        for (int ni = 0; ni < 4; ni++)
          acc[mi][ni] = wmma_f16(af[mi], bf[ni], acc[mi][ni]);
    }

    if (has_next) {
      storeL(cur ^ 1);
      __syncthreads();
      cur ^= 1;
    }
  }

  const float sd = scales[2];
  const int nlane = lane & 15;
  const int mlane = (lane >> 4) * 8;
  #pragma unroll
  for (int mi = 0; mi < 2; mi++)
    #pragma unroll
    for (int ni = 0; ni < 4; ni++) {
      long mbase = m0 + waveM * 32 + mi * 16 + mlane;
      int  nn = n0 + waveN * 64 + ni * 16 + nlane;
      #pragma unroll
      for (int e = 0; e < 8; e++)
        Out[(mbase + e) * (long)Nd + nn] = sd * acc[mi][ni][e];
    }
}

extern "C" void kernel_launch(void* const* d_in, const int* in_sizes, int n_in,
                              void* d_out, int out_size, void* d_ws, size_t ws_size,
                              hipStream_t stream) {
  const float* x  = (const float*)d_in[0];   // [8,4096,768]
  const float* wg = (const float*)d_in[1];   // [8,96,256]
  const float* wu = (const float*)d_in[2];   // [8,96,256]
  const float* wd = (const float*)d_in[3];   // [8,256,96]
  float* out = (float*)d_out;                // [8,4096,768] f32

  const long M = 32768;          // 8*4096 tokens
  const int  Dd = 768, Hh = 2048;
  const int  WN = 8 * 96 * 256;  // elements per weight tensor

  char* ws = (char*)d_ws;
  size_t off = 0;
  auto carve = [&](size_t bytes) { size_t p = off; off = (off + bytes + 255) & ~(size_t)255; return p; };
  float*    scales = (float*)   (ws + carve(16));
  _Float16* xh     = (_Float16*)(ws + carve((size_t)M * Dd * 2));   // 50.3 MB
  _Float16* bgT    = (_Float16*)(ws + carve((size_t)Hh * Dd * 2));  //  3.1 MB
  _Float16* buT    = (_Float16*)(ws + carve((size_t)Hh * Dd * 2));  //  3.1 MB
  _Float16* bdT    = (_Float16*)(ws + carve((size_t)Dd * Hh * 2));  //  3.1 MB
  _Float16* hbuf   = (_Float16*)(ws + carve((size_t)M * Hh * 2));   // 134 MB

  oct_absmean<<<3, 256, 0, stream>>>(wg, wu, wd, scales, WN);
  oct_cvt_f16<<<(int)(M * Dd / 4 / 256), 256, 0, stream>>>(x, xh);
  int bigBlocks = (int)(((long)Hh * Dd) / 256);
  oct_build_bigT<<<bigBlocks, 256, 0, stream>>>(wg, scales, 0, bgT, 96, 256);
  oct_build_bigT<<<bigBlocks, 256, 0, stream>>>(wu, scales, 1, buT, 96, 256);
  oct_build_bigT<<<bigBlocks, 256, 0, stream>>>(wd, scales, 2, bdT, 256, 96);
  oct_gemm_gateup<<<dim3(Hh / 128, (int)(M / 128)), 256, 0, stream>>>(xh, bgT, buT, scales, hbuf);
  oct_gemm_down<<<dim3(Dd / 128, (int)(M / 128)), 256, 0, stream>>>(hbuf, bdT, scales, out);
}
